// lstm_52587579572675
// MI455X (gfx1250) — matrix operand order repaired and checked by reading, not compiled, not run on hardware
//
#include <hip/hip_runtime.h>
#include <hip/hip_bf16.h>
#include <math.h>

#define NWG   32
#define TPB   256
#define LSEQ  256
#define BSZ   128
#define HS    512
#define NGATE (4 * HS)
#define HBUF  ((size_t)BSZ * HS)   // elements in one hidden-state buffer

typedef __bf16 bf16;
typedef bf16  v8bf  __attribute__((ext_vector_type(8)));
typedef bf16  v16bf __attribute__((ext_vector_type(16)));
typedef float v8f   __attribute__((ext_vector_type(8)));

__device__ inline float sigm(float v)      { return 1.0f / (1.0f + __expf(-v)); }
__device__ inline float tanh_fast(float v) { return 1.0f - 2.0f / (__expf(2.0f * v) + 1.0f); }

// A-fragment (16x32 bf16): lane m=l16, half=lh -> K [lh*16, lh*16+16) contiguous
__device__ inline v16bf load_afrag(const bf16* __restrict__ arow, int kb, int lh) {
  return *(const v16bf*)(arow + kb + lh * 16);
}

// z[:, {i,f,g,o} x 16 cols] += A[128x512] @ W^T-packed  (B frag: lane n=l16, K=kb+lh*16..+16 contiguous)
__device__ inline void gemm4(v8f acc[4], const bf16* __restrict__ A,
                             const bf16* __restrict__ Bt, int row_a, int jcol, int lh) {
  const bf16* arow = A + (size_t)row_a * HS;
  #pragma unroll 4
  for (int kb = 0; kb < HS; kb += 32) {
    v16bf af = load_afrag(arow, kb, lh);
    #pragma unroll
    for (int g = 0; g < 4; ++g) {
      const bf16* bp = Bt + ((size_t)(g * HS + jcol)) * HS + kb + lh * 16;
      v16bf bfr = *(const v16bf*)bp;
      acc[g] = __builtin_amdgcn_wmma_f32_16x16x32_bf16(false, af, false, bfr,
                                                       (short)0, acc[g], false, false);
    }
  }
}

__device__ inline v8f gemm1(v8f acc, const bf16* __restrict__ A,
                            const bf16* __restrict__ Bt, int row_a, int ncol, int lh) {
  const bf16* arow = A + (size_t)row_a * HS;
  const bf16* bcol = Bt + (size_t)ncol * HS;
  #pragma unroll 4
  for (int kb = 0; kb < HS; kb += 32) {
    v16bf af  = load_afrag(arow, kb, lh);
    v16bf bfr = *(const v16bf*)(bcol + kb + lh * 16);
    acc = __builtin_amdgcn_wmma_f32_16x16x32_bf16(false, af, false, bfr,
                                                  (short)0, acc, false, false);
  }
  return acc;
}

// device-wide split barrier: monotonic counter, one arrive per WG
__device__ inline void grid_barrier(unsigned* cnt, unsigned target) {
  __syncthreads();
  if (threadIdx.x == 0) {
    __threadfence();                                   // release h stores to device scope
    __hip_atomic_fetch_add(cnt, 1u, __ATOMIC_RELEASE, __HIP_MEMORY_SCOPE_AGENT);
    while (__hip_atomic_load(cnt, __ATOMIC_ACQUIRE, __HIP_MEMORY_SCOPE_AGENT) < target)
      __builtin_amdgcn_s_sleep(1);
  }
  __syncthreads();
  __threadfence();                                     // acquire for all waves in the WG
}

__global__ void init_ws(unsigned* cnt, bf16* h0, bf16* h1) {
  const int i = blockIdx.x * blockDim.x + threadIdx.x;
  if (i == 0) *cnt = 0u;
  if (i < BSZ * HS) { h0[i] = (bf16)0.f; h1[i] = (bf16)0.f; }   // zeroes buffer half 0
}

// Wh0 / Wx1 / Wh1: fp32 [K=512][N=2048] -> bf16 [N][K]
__global__ void pack_big(const float* __restrict__ Wh0, const float* __restrict__ Wx1,
                         const float* __restrict__ Wh1,
                         bf16* __restrict__ Wh0t, bf16* __restrict__ Wx1t, bf16* __restrict__ Wh1t) {
  const long idx = (long)blockIdx.x * blockDim.x + threadIdx.x;   // 3 * 2^20
  const int  which = (int)(idx >> 20);
  const long r = idx & ((1L << 20) - 1);
  const int  k = (int)(r & (HS - 1));
  const int  n = (int)(r >> 9);
  const float* s = (which == 0) ? Wh0 : ((which == 1) ? Wx1 : Wh1);
  bf16*        d = (which == 0) ? Wh0t : ((which == 1) ? Wx1t : Wh1t);
  d[(size_t)n * HS + k] = (bf16)s[(size_t)k * NGATE + n];
}

// Wr1 / Wi1: fp32 [512][10] -> bf16 [16][512], cols 10..15 zero-padded
__global__ void pack_mlp(const float* __restrict__ Wr1, const float* __restrict__ Wi1,
                         bf16* __restrict__ Wr1t, bf16* __restrict__ Wi1t) {
  const int idx = blockIdx.x * blockDim.x + threadIdx.x;          // 2 * 16 * 512
  const int which = idx >> 13;
  const int r = idx & 8191;
  const int k = r & (HS - 1);
  const int n = r >> 9;
  const float* s = which ? Wi1 : Wr1;
  bf16*        d = which ? Wi1t : Wr1t;
  d[(size_t)n * HS + k] = (n < 10) ? (bf16)s[(size_t)k * 10 + n] : (bf16)0.f;
}

__global__ void __launch_bounds__(TPB, 1)
lstm_main(const float* __restrict__ x,   const float* __restrict__ Wx0,
          const float* __restrict__ b0,  const float* __restrict__ b1,
          const float* __restrict__ br1, const float* __restrict__ Wr2, const float* __restrict__ br2,
          const float* __restrict__ bi1, const float* __restrict__ Wi2, const float* __restrict__ bi2,
          const bf16* __restrict__ Wh0t, const bf16* __restrict__ Wx1t, const bf16* __restrict__ Wh1t,
          const bf16* __restrict__ Wr1t, const bf16* __restrict__ Wi1t,
          bf16* __restrict__ h0base,     // 2 contiguous [B][H] buffers (double buffer)
          bf16* __restrict__ h1base,     // 2 contiguous [B][H] buffers (double buffer)
          bf16* __restrict__ h0r, bf16* __restrict__ h1r,
          unsigned* cnt, float* __restrict__ out)
{
  __shared__ float ldsR[8][16][16];
  __shared__ float ldsI[8][16][16];

  const int wg   = blockIdx.x;
  const int wave = threadIdx.x >> 5;
  const int lane = threadIdx.x & 31;
  const int lh   = lane >> 4;           // lane half
  const int l16  = lane & 15;

  const int jcol  = wg * 16 + l16;      // hidden column owned by this lane
  const int mbase = wave * 16;          // batch-row tile of this wave
  const int row_a = mbase + l16;        // A-fragment row
  const int rowc0 = mbase + 8 * lh;     // first C/D row for this lane

  float b0g[4], b1g[4], w0g[4], w1g[4];
  #pragma unroll
  for (int g = 0; g < 4; ++g) {
    b0g[g] = b0[g * HS + jcol];
    b1g[g] = b1[g * HS + jcol];
    w0g[g] = Wx0[0 * NGATE + g * HS + jcol];   // one-hot row 0 (x = -1)
    w1g[g] = Wx0[1 * NGATE + g * HS + jcol];   // one-hot row 1 (x = +1)
  }
  float c0r[8], c1r[8];
  #pragma unroll
  for (int r = 0; r < 8; ++r) { c0r[r] = 0.f; c1r[r] = 0.f; }

  unsigned bar = 0;

  for (int t = 0; t < LSEQ; ++t) {
    // uniform scalar offsets select the double-buffer half -> SGPR base + voffset (global, not flat)
    const size_t po = (size_t)(t & 1) * HBUF;
    const size_t qo = (size_t)((t & 1) ^ 1) * HBUF;

    // -------- layer 0: z = onehot(x_t)@Wx0 + h0@Wh0 + b0 --------
    v8f acc[4] = {};
    gemm4(acc, h0base + po, Wh0t, row_a, jcol, lh);
    #pragma unroll
    for (int r = 0; r < 8; ++r) {
      const int   row = rowc0 + r;
      const float xv  = x[(size_t)row * LSEQ + t];
      const float zi = acc[0][r] + b0g[0] + (xv > 0.f ? w1g[0] : w0g[0]);
      const float zf = acc[1][r] + b0g[1] + (xv > 0.f ? w1g[1] : w0g[1]);
      const float zg = acc[2][r] + b0g[2] + (xv > 0.f ? w1g[2] : w0g[2]);
      const float zo = acc[3][r] + b0g[3] + (xv > 0.f ? w1g[3] : w0g[3]);
      const float cn = sigm(zf) * c0r[r] + sigm(zi) * tanh_fast(zg);
      c0r[r] = cn;
      const float hv = sigm(zo) * tanh_fast(cn);
      const size_t hidx = (size_t)row * HS + jcol;
      h0base[qo + hidx] = (bf16)hv;
      h0r[hidx]         = (bf16)fmaxf(hv, 0.f);
    }
    grid_barrier(cnt, (++bar) * NWG);

    // -------- layer 1: z = relu(h0_new)@Wx1 + h1@Wh1 + b1 --------
    v8f acc2[4] = {};
    gemm4(acc2, h0r,         Wx1t, row_a, jcol, lh);
    gemm4(acc2, h1base + po, Wh1t, row_a, jcol, lh);
    #pragma unroll
    for (int r = 0; r < 8; ++r) {
      const int   row = rowc0 + r;
      const float zi = acc2[0][r] + b1g[0];
      const float zf = acc2[1][r] + b1g[1];
      const float zg = acc2[2][r] + b1g[2];
      const float zo = acc2[3][r] + b1g[3];
      const float cn = sigm(zf) * c1r[r] + sigm(zi) * tanh_fast(zg);
      c1r[r] = cn;
      const float hv = sigm(zo) * tanh_fast(cn);
      const size_t hidx = (size_t)row * HS + jcol;
      h1base[qo + hidx] = (bf16)hv;
      h1r[hidx]         = (bf16)fmaxf(hv, 0.f);
    }
    grid_barrier(cnt, (++bar) * NWG);

    // -------- real_to_complex MLP + output (WG0 only) --------
    if (blockIdx.x == 0) {
      v8f ar = {}, ai = {};
      ar = gemm1(ar, h1r, Wr1t, row_a, l16, lh);
      ai = gemm1(ai, h1r, Wi1t, row_a, l16, lh);
      const float brc = (l16 < 10) ? br1[l16] : 0.f;
      const float bic = (l16 < 10) ? bi1[l16] : 0.f;
      #pragma unroll
      for (int r = 0; r < 8; ++r) {
        ldsR[wave][8 * lh + r][l16] = fmaxf(ar[r] + brc, 0.f);
        ldsI[wave][8 * lh + r][l16] = fmaxf(ai[r] + bic, 0.f);
      }
      __syncthreads();
      float a0, a1;
      if (lh == 0) {                                   // magnitude path
        a0 = br2[0]; a1 = br2[1];
        #pragma unroll
        for (int j = 0; j < 10; ++j) {
          const float v = ldsR[wave][l16][j];
          a0 += v * Wr2[j * 2 + 0];
          a1 += v * Wr2[j * 2 + 1];
        }
      } else {                                         // phase path
        a0 = bi2[0]; a1 = bi2[1];
        #pragma unroll
        for (int j = 0; j < 10; ++j) {
          const float v = ldsI[wave][l16][j];
          a0 += v * Wi2[j * 2 + 0];
          a1 += v * Wi2[j * 2 + 1];
        }
        a0 = 3.14159265358979f * a0 / (1.f + fabsf(a0));  // pi * softsign
        a1 = 3.14159265358979f * a1 / (1.f + fabsf(a1));
      }
      const float p0 = __shfl_xor(a0, 16, 32);
      const float p1 = __shfl_xor(a1, 16, 32);
      if (lh == 0) {
        const int b = mbase + l16;
        float4 o;
        o.x = a0 * __cosf(p0); o.y = a0 * __sinf(p0);
        o.z = a1 * __cosf(p1); o.w = a1 * __sinf(p1);
        *(float4*)(out + ((size_t)b * LSEQ + t) * 4) = o;   // complex64 [B][L][2]
      }
      __syncthreads();
    }
  }
}

extern "C" void kernel_launch(void* const* d_in, const int* in_sizes, int n_in,
                              void* d_out, int out_size, void* d_ws, size_t ws_size,
                              hipStream_t stream) {
  const float* x   = (const float*)d_in[0];
  const float* Wx0 = (const float*)d_in[1];
  const float* Wh0 = (const float*)d_in[2];
  const float* b0  = (const float*)d_in[3];
  const float* Wx1 = (const float*)d_in[4];
  const float* Wh1 = (const float*)d_in[5];
  const float* b1  = (const float*)d_in[6];
  const float* Wr1 = (const float*)d_in[7];
  const float* br1 = (const float*)d_in[8];
  const float* Wr2 = (const float*)d_in[9];
  const float* br2 = (const float*)d_in[10];
  const float* Wi1 = (const float*)d_in[11];
  const float* bi1 = (const float*)d_in[12];
  const float* Wi2 = (const float*)d_in[13];
  const float* bi2 = (const float*)d_in[14];
  (void)in_sizes; (void)n_in; (void)out_size; (void)ws_size;

  char* ws = (char*)d_ws;
  size_t off = 0;
  auto carve = [&](size_t bytes) -> char* {
    char* pp = ws + off;
    off += (bytes + 255) & ~(size_t)255;
    return pp;
  };
  unsigned* cnt  = (unsigned*)carve(256);
  bf16* Wh0t   = (bf16*)carve((size_t)NGATE * HS * 2);
  bf16* Wx1t   = (bf16*)carve((size_t)NGATE * HS * 2);
  bf16* Wh1t   = (bf16*)carve((size_t)NGATE * HS * 2);
  bf16* Wr1t   = (bf16*)carve((size_t)16 * HS * 2);
  bf16* Wi1t   = (bf16*)carve((size_t)16 * HS * 2);
  bf16* h0base = (bf16*)carve(2 * HBUF * 2);   // double buffer, contiguous
  bf16* h1base = (bf16*)carve(2 * HBUF * 2);   // double buffer, contiguous
  bf16* h0r    = (bf16*)carve(HBUF * 2);
  bf16* h1r    = (bf16*)carve(HBUF * 2);

  init_ws <<<256,   TPB, 0, stream>>>(cnt, h0base, h1base);
  pack_big<<<12288, TPB, 0, stream>>>(Wh0, Wx1, Wh1, Wh0t, Wx1t, Wh1t);
  pack_mlp<<<64,    TPB, 0, stream>>>(Wr1, Wi1, Wr1t, Wi1t);
  lstm_main<<<NWG,  TPB, 0, stream>>>(x, Wx0, b0, b1, br1, Wr2, br2, bi1, Wi2, bi2,
                                      Wh0t, Wx1t, Wh1t, Wr1t, Wi1t,
                                      h0base, h1base, h0r, h1r,
                                      cnt, (float*)d_out);
}
